// Fuse_75136157876258
// MI455X (gfx1250) — matrix operand order repaired and checked by reading, not compiled, not run on hardware
//
#include <hip/hip_runtime.h>

typedef __bf16 bf16;
typedef __attribute__((ext_vector_type(16))) __bf16 v16bf;
typedef __attribute__((ext_vector_type(8)))  __bf16 v8bf;
typedef __attribute__((ext_vector_type(8)))  float   v8f;
typedef __attribute__((ext_vector_type(4)))  float   v4f;

#define LDA2 72         // padded LDS row stride (bf16 elems), 144B = 36 banks -> conflict-free
#define MROWS 8192      // B*S = 64*128
#define NOUT 1024

// ---------------- small utility kernels ----------------

__global__ __launch_bounds__(256) void zero_f32(float* __restrict__ p, int n) {
    int i = blockIdx.x * 256 + threadIdx.x;
    if (i < n) p[i] = 0.f;
}

__global__ __launch_bounds__(256) void f32_to_bf16(const float* __restrict__ s,
                                                   bf16* __restrict__ d, int n) {
    int i = blockIdx.x * 256 + threadIdx.x;
    if (i < n) d[i] = (bf16)s[i];
}

// idx24 = FPS_2[b][FPS_3]; idx14 = FPS_1[b][idx24]; idx04 = FPS_0[b][idx14]
__global__ __launch_bounds__(256) void chain_idx(const int* __restrict__ F0,
                                                 const int* __restrict__ F1,
                                                 const int* __restrict__ F2,
                                                 const int* __restrict__ F3,
                                                 int* __restrict__ i24,
                                                 int* __restrict__ i14,
                                                 int* __restrict__ i04) {
    int r = blockIdx.x * 256 + threadIdx.x;   // 8192 threads
    int b = r >> 7;
    int a3  = F3[r];                 // [0,256)
    int a24 = F2[b * 256  + a3];     // [0,512)
    int a14 = F1[b * 512  + a24];    // [0,1024)
    int a04 = F0[b * 1024 + a14];    // [0,2048)
    i24[r] = a24; i14[r] = a14; i04[r] = a04;
}

// ---------------- fused gather + GEMM (bf16 WMMA) + BN-stat epilogue ----------------
// Y[r][n] = sum_k X[b, row(r), k] * W[n][k] + bias[n];  stats += {sum_r Y, sum_r Y^2}
// Workgroup: 256 thr = 8 waves; tile M=128, N=64, K-step 64; wave tile 32x32.
// B (weights, bf16) staged via global_load_async_to_lds_b128 (ASYNCcnt);
// A staged via VGPRs (gather + f32->bf16 convert) into LDS.
__global__ __launch_bounds__(256)
void gemm_bn(const float* __restrict__ X, const bf16* __restrict__ Wb,
             const float* __restrict__ bias, const int* __restrict__ idx,
             int srcN, int K,
             float* __restrict__ Y, float* __restrict__ stats /* [2048] */) {
    __shared__ bf16 Al[128 * LDA2];
    __shared__ bf16 Bl[64 * LDA2];
    __shared__ float sSum[64], sSq[64];

    const int tid   = threadIdx.x;
    const int mTile = blockIdx.y * 128;
    const int nTile = blockIdx.x * 64;
    if (tid < 64) { sSum[tid] = 0.f; sSq[tid] = 0.f; }

    // A-tile cooperative load: 128 rows x 64 k, 32 f32 per thread (gather + cvt)
    const int arow  = tid >> 1;
    const int akoff = (tid & 1) << 5;          // 0 or 32
    const int r = mTile + arow;
    const int b = r >> 7;
    const int s = r & 127;
    const int srow = idx ? idx[r] : s;
    const float* aptr = X + ((size_t)b * srcN + srow) * K + akoff;

    // B-tile async load: 64 rows (cout) x 64 k, 2x16B per thread
    const int brow  = tid >> 2;
    const int bkoff = (tid & 3) << 4;          // 0,16,32,48
    const bf16* bptr = Wb + (size_t)(nTile + brow) * K + bkoff;
    // low 32 bits of a generic shared pointer == wave-relative LDS byte offset
    const unsigned ldsB0 = (unsigned)(size_t)&Bl[brow * LDA2 + bkoff];
    const unsigned ldsB1 = ldsB0 + 16u;

    const int lane = tid & 31;
    const int wave = tid >> 5;
    const int wmb  = (wave >> 1) << 5;   // wave M base: 0,32,64,96
    const int wnb  = (wave & 1)  << 5;   // wave N base: 0,32
    const int lm   = lane & 15;
    const int half = lane >> 4;

    v8f acc[2][2];
#pragma unroll
    for (int i = 0; i < 2; i++)
#pragma unroll
        for (int j = 0; j < 2; j++)
#pragma unroll
            for (int e = 0; e < 8; e++) acc[i][j][e] = 0.f;

    for (int k0 = 0; k0 < K; k0 += 64) {
        __syncthreads();
        {   // B tile: cache -> LDS direct (no VGPR round trip), ASYNCcnt-tracked
            unsigned long long ga0 = (unsigned long long)(bptr + k0);
            unsigned long long ga1 = ga0 + 16ull;
            asm volatile("global_load_async_to_lds_b128 %0, %2, off\n\t"
                         "global_load_async_to_lds_b128 %1, %3, off"
                         :: "v"(ldsB0), "v"(ldsB1), "v"(ga0), "v"(ga1)
                         : "memory");
        }
        {   // A tile: gather + f32 -> bf16 -> LDS
            const float* p = aptr + k0;
            v4f x[8];
#pragma unroll
            for (int u = 0; u < 8; u++) x[u] = *(const v4f*)(p + 4 * u);
            if (k0 + 64 < K) __builtin_prefetch(p + 64, 0, 1);
#pragma unroll
            for (int q = 0; q < 4; q++) {
                v8bf o;
#pragma unroll
                for (int e = 0; e < 8; e++)
                    o[e] = (bf16)x[2 * q + (e >> 2)][e & 3];
                *(v8bf*)&Al[arow * LDA2 + akoff + 8 * q] = o;
            }
        }
        asm volatile("s_wait_asynccnt 0x0" ::: "memory");
        __syncthreads();

#pragma unroll
        for (int kk = 0; kk < 64; kk += 32) {
            // A frag: lane m=lm, half h: elems 0..7 = K{8h..}, 8..15 = K{16+8h..}
            v16bf af[2], bfm[2];
#pragma unroll
            for (int i = 0; i < 2; i++) {
                const bf16* base = &Al[(wmb + i * 16 + lm) * LDA2 + kk];
                v8bf lo = *(const v8bf*)(base + half * 8);
                v8bf hi = *(const v8bf*)(base + 16 + half * 8);
                af[i] = __builtin_shufflevector(lo, hi, 0,1,2,3,4,5,6,7,8,9,10,11,12,13,14,15);
            }
            // B frag: lane col n=lm, half h: elems 0..7 = K{8h..}, 8..15 = K{16+8h..}
#pragma unroll
            for (int j = 0; j < 2; j++) {
                const bf16* base = &Bl[(wnb + j * 16 + lm) * LDA2 + kk];
                v8bf lo = *(const v8bf*)(base + half * 8);
                v8bf hi = *(const v8bf*)(base + 16 + half * 8);
                bfm[j] = __builtin_shufflevector(lo, hi, 0,1,2,3,4,5,6,7,8,9,10,11,12,13,14,15);
            }
#pragma unroll
            for (int i = 0; i < 2; i++)
#pragma unroll
                for (int j = 0; j < 2; j++)
                    acc[i][j] = __builtin_amdgcn_wmma_f32_16x16x32_bf16(
                        false, af[i], false, bfm[j], (short)0, acc[i][j], false, false);
        }
    }

    // epilogue: +bias, store Y, per-channel partial sums for BN
#pragma unroll
    for (int j = 0; j < 2; j++) {
        const int colLocal = wnb + j * 16 + lm;
        const int col = nTile + colLocal;
        const float bv = bias[col];
        float ps = 0.f, pq = 0.f;
#pragma unroll
        for (int i = 0; i < 2; i++) {
            const int mrow = mTile + wmb + i * 16 + half * 8;
#pragma unroll
            for (int v = 0; v < 8; v++) {
                float y = acc[i][j][v] + bv;
                Y[(size_t)(mrow + v) * NOUT + col] = y;
                ps += y; pq += y * y;
            }
        }
        atomicAdd(&sSum[colLocal], ps);
        atomicAdd(&sSq[colLocal], pq);
    }
    __syncthreads();
    if (tid < 64) {
        atomicAdd(&stats[nTile + tid], sSum[tid]);
        atomicAdd(&stats[NOUT + nTile + tid], sSq[tid]);
    }
}

// ---------------- BN stat finalize: (sum,sumsq) -> (scale,shift) in place ----------------
__global__ __launch_bounds__(256) void finalize_stats(float* __restrict__ st,
                                                      const float* __restrict__ g,
                                                      const float* __restrict__ be) {
    int c = blockIdx.x * 256 + threadIdx.x;       // 1024
    const float inv = 1.f / 8192.f;
    float mu  = st[c] * inv;
    float var = st[NOUT + c] * inv - mu * mu;
    float sc  = rsqrtf(var + 1e-5f) * g[c];
    st[c]        = sc;
    st[NOUT + c] = be[c] - mu * sc;
}

__device__ inline v4f bn_lrelu4(const float* __restrict__ y, const float* __restrict__ st,
                                int base, int c) {
    v4f v  = *(const v4f*)(y + base);
    v4f sc = *(const v4f*)(st + c);
    v4f sh = *(const v4f*)(st + NOUT + c);
    v4f r;
#pragma unroll
    for (int e = 0; e < 4; e++) {
        float t = v[e] * sc[e] + sh[e];
        r[e] = (t >= 0.f) ? t : 0.2f * t;
    }
    return r;
}

// s = f4 + sum_i bn_lrelu(y_i)
__global__ __launch_bounds__(256)
void combine(const float* __restrict__ f4,
             const float* __restrict__ y0, const float* __restrict__ y1,
             const float* __restrict__ y2, const float* __restrict__ y3,
             const float* __restrict__ st0, const float* __restrict__ st1,
             const float* __restrict__ st2, const float* __restrict__ st3,
             float* __restrict__ S) {
    int i = blockIdx.x * 256 + threadIdx.x;       // 2,097,152 float4s
    int base = i * 4;
    int c = base & (NOUT - 1);
    v4f acc = *(const v4f*)(f4 + base);
    v4f a0 = bn_lrelu4(y0, st0, base, c);
    v4f a1 = bn_lrelu4(y1, st1, base, c);
    v4f a2 = bn_lrelu4(y2, st2, base, c);
    v4f a3 = bn_lrelu4(y3, st3, base, c);
#pragma unroll
    for (int e = 0; e < 4; e++) acc[e] += a0[e] + a1[e] + a2[e] + a3[e];
    *(v4f*)(S + base) = acc;
}

// f4_new = bn_lrelu(y4) + f4
__global__ __launch_bounds__(256)
void final_add(const float* __restrict__ y4, const float* __restrict__ st,
               const float* __restrict__ f4, float* __restrict__ out) {
    int i = blockIdx.x * 256 + threadIdx.x;
    int base = i * 4;
    int c = base & (NOUT - 1);
    v4f a = bn_lrelu4(y4, st, base, c);
    v4f f = *(const v4f*)(f4 + base);
#pragma unroll
    for (int e = 0; e < 4; e++) a[e] += f[e];
    *(v4f*)(out + base) = a;
}

// ---------------- host launch ----------------
extern "C" void kernel_launch(void* const* d_in, const int* in_sizes, int n_in,
                              void* d_out, int out_size, void* d_ws, size_t ws_size,
                              hipStream_t stream) {
    // input order: num_point, f0..f4, FPS_0..FPS_3, then (w,b,g,be) x {04,14,24,34,4}
    const float* f0 = (const float*)d_in[1];
    const float* f1 = (const float*)d_in[2];
    const float* f2 = (const float*)d_in[3];
    const float* f3 = (const float*)d_in[4];
    const float* f4 = (const float*)d_in[5];
    const int* F0 = (const int*)d_in[6];
    const int* F1 = (const int*)d_in[7];
    const int* F2 = (const int*)d_in[8];
    const int* F3 = (const int*)d_in[9];
    const float *w04 = (const float*)d_in[10], *b04 = (const float*)d_in[11],
                *g04 = (const float*)d_in[12], *be04 = (const float*)d_in[13];
    const float *w14 = (const float*)d_in[14], *b14 = (const float*)d_in[15],
                *g14 = (const float*)d_in[16], *be14 = (const float*)d_in[17];
    const float *w24 = (const float*)d_in[18], *b24 = (const float*)d_in[19],
                *g24 = (const float*)d_in[20], *be24 = (const float*)d_in[21];
    const float *w34 = (const float*)d_in[22], *b34 = (const float*)d_in[23],
                *g34 = (const float*)d_in[24], *be34 = (const float*)d_in[25];
    const float *w4  = (const float*)d_in[26], *b4  = (const float*)d_in[27],
                *g4  = (const float*)d_in[28], *be4 = (const float*)d_in[29];

    char* ws = (char*)d_ws;
    int*   i24   = (int*)(ws + 0);
    int*   i14   = (int*)(ws + 32768);
    int*   i04   = (int*)(ws + 65536);
    float* stats = (float*)(ws + 98304);           // 5 blocks * 2048 f32
    bf16*  wb    = (bf16*)(ws + 139264);           // bf16 weights
    bf16*  w04b = wb;                               //  65536 elems
    bf16*  w14b = wb + 65536;                       // 131072
    bf16*  w24b = wb + 196608;                      // 262144
    bf16*  w34b = wb + 458752;                      // 524288
    bf16*  w4b  = wb + 983040;                      // 1048576 -> ends at 2031616
    float* y04  = (float*)(ws + 4202496);
    float* y14  = y04 + 8388608;
    float* y24  = y14 + 8388608;
    float* y34  = y24 + 8388608;
    float* Sbuf = y34 + 8388608;
    float* y4b  = Sbuf + 8388608;

    float* out = (float*)d_out;
    const size_t SEC = 8388608;   // each output tensor, flat f32 elems

    zero_f32<<<(5 * 2048 + 255) / 256, 256, 0, stream>>>(stats, 5 * 2048);
    chain_idx<<<32, 256, 0, stream>>>(F0, F1, F2, F3, i24, i14, i04);
    f32_to_bf16<<<(65536   + 255) / 256, 256, 0, stream>>>(w04, w04b, 65536);
    f32_to_bf16<<<(131072  + 255) / 256, 256, 0, stream>>>(w14, w14b, 131072);
    f32_to_bf16<<<(262144  + 255) / 256, 256, 0, stream>>>(w24, w24b, 262144);
    f32_to_bf16<<<(524288  + 255) / 256, 256, 0, stream>>>(w34, w34b, 524288);
    f32_to_bf16<<<(1048576 + 255) / 256, 256, 0, stream>>>(w4,  w4b,  1048576);

    dim3 gg(NOUT / 64, MROWS / 128);   // (16, 64)
    gemm_bn<<<gg, 256, 0, stream>>>(f0, w04b, b04, i04, 2048, 64,   y04, stats + 0);
    gemm_bn<<<gg, 256, 0, stream>>>(f1, w14b, b14, i14, 1024, 128,  y14, stats + 2048);
    gemm_bn<<<gg, 256, 0, stream>>>(f2, w24b, b24, i24, 512,  256,  y24, stats + 4096);
    gemm_bn<<<gg, 256, 0, stream>>>(f3, w34b, b34, F3,  256,  512,  y34, stats + 6144);

    finalize_stats<<<4, 256, 0, stream>>>(stats + 0,    g04, be04);
    finalize_stats<<<4, 256, 0, stream>>>(stats + 2048, g14, be14);
    finalize_stats<<<4, 256, 0, stream>>>(stats + 4096, g24, be24);
    finalize_stats<<<4, 256, 0, stream>>>(stats + 6144, g34, be34);

    combine<<<8192, 256, 0, stream>>>(f4, y04, y14, y24, y34,
                                      stats + 0, stats + 2048, stats + 4096, stats + 6144,
                                      Sbuf);

    gemm_bn<<<gg, 256, 0, stream>>>(Sbuf, w4b, b4, nullptr, 128, 1024, y4b, stats + 8192);
    finalize_stats<<<4, 256, 0, stream>>>(stats + 8192, g4, be4);
    final_add<<<8192, 256, 0, stream>>>(y4b, stats + 8192, f4, out + 4 * SEC);

    // pass-through outputs f0..f3
    hipMemcpyAsync(out + 0 * SEC, f0, SEC * sizeof(float), hipMemcpyDeviceToDevice, stream);
    hipMemcpyAsync(out + 1 * SEC, f1, SEC * sizeof(float), hipMemcpyDeviceToDevice, stream);
    hipMemcpyAsync(out + 2 * SEC, f2, SEC * sizeof(float), hipMemcpyDeviceToDevice, stream);
    hipMemcpyAsync(out + 3 * SEC, f3, SEC * sizeof(float), hipMemcpyDeviceToDevice, stream);
}
